// StudentModel_47038481825984
// MI455X (gfx1250) — matrix operand order repaired and checked by reading, not compiled, not run on hardware
//
#include <hip/hip_runtime.h>
#include <math.h>

typedef __attribute__((ext_vector_type(16))) _Float16 v16h;
typedef __attribute__((ext_vector_type(8)))  _Float16 v8h;
typedef __attribute__((ext_vector_type(8)))  float    v8f;

__device__ __forceinline__ v8f wmma_f16(v16h a, v16h b, v8f c) {
    return __builtin_amdgcn_wmma_f32_16x16x32_f16(
        false, a, false, b, (short)0, c, false, false);
}

__device__ __forceinline__ v16h cmb(v8h lo, v8h hi) {
    return __builtin_shufflevector(lo, hi, 0, 1, 2, 3, 4, 5, 6, 7,
                                   8, 9, 10, 11, 12, 13, 14, 15);
}

__device__ __forceinline__ float gelu_exact(float x) {
    return 0.5f * x * (1.0f + erff(x * 0.70710678118654752f));
}

// ---------------------------------------------------------------------------
// One-time f32 -> f16 weight conversion (plain).
// ---------------------------------------------------------------------------
__global__ void cvt_f32_to_f16(const float* __restrict__ in, _Float16* __restrict__ out, int n) {
    int i = blockIdx.x * 2048 + threadIdx.x * 8;
#pragma unroll
    for (int j = 0; j < 8; ++j)
        if (i + j < n) out[i + j] = (_Float16)in[i + j];
}

// Conv weight convert + K-permute: out[m][dk*512+ci] = in[m][ci][dk]  (f16)
__global__ void conv_w_perm_f16(const float* __restrict__ in, _Float16* __restrict__ out, int k) {
    int Ktot = 512 * k;
    int idx = blockIdx.x * 256 + threadIdx.x;
    if (idx >= 512 * Ktot) return;
    int m = idx / Ktot, r = idx - m * Ktot;
    int dk = r >> 9, ci = r & 511;
    out[idx] = (_Float16)in[((size_t)m * 512 + ci) * k + dk];
}

// ---------------------------------------------------------------------------
// Conv layer 1: cin=1, k=10, s=5. Fused channel-LN(ddof=1)+GELU.
// Output token-major f16 [B][16000][512]. Grid: (1000, 8), block 256.
// ---------------------------------------------------------------------------
__global__ void conv1_ln_gelu(const float* __restrict__ x,
                              const float* __restrict__ w,
                              const float* __restrict__ bias,
                              const float* __restrict__ nw,
                              const float* __restrict__ nb,
                              _Float16* __restrict__ out) {
    const int bidx = blockIdx.y;
    const int t0   = blockIdx.x * 16;
    const int tid  = threadIdx.x;
    __shared__ float wsm[5120];
    __shared__ float hs[512][16];
    __shared__ float rsum[16][16], rsq[16][16];
    __shared__ float smean[16], sinv[16];

    for (int i = tid; i < 5120; i += 256) wsm[i] = w[i];
    __syncthreads();

    {   // compute: thread -> (token n, 32-channel group g)
        const int n = tid & 15, g = tid >> 4;
        const int t = t0 + n;
        const float* xb = x + (size_t)bidx * 80000;
        const int p0 = 5 * t - 9;
        float xv[10];
#pragma unroll
        for (int dk = 0; dk < 10; ++dk) {
            int p = p0 + dk;
            xv[dk] = (p >= 0 && p < 80000) ? xb[p] : 0.0f;
        }
        for (int mi = 0; mi < 32; ++mi) {
            int m = g * 32 + mi;
            float acc = bias[m];
#pragma unroll
            for (int dk = 0; dk < 10; ++dk) acc += xv[dk] * wsm[m * 10 + dk];
            hs[m][n] = acc;
        }
    }
    __syncthreads();
    // LN epilogue: thread -> (token tk, channel group cg)
    const int tk = tid >> 4, cg = tid & 15;
    float ssum = 0.f, ssq = 0.f;
    for (int j = 0; j < 32; ++j) { float v = hs[cg * 32 + j][tk]; ssum += v; ssq += v * v; }
    rsum[tk][cg] = ssum; rsq[tk][cg] = ssq;
    __syncthreads();
    if (tid < 16) {
        float ts = 0.f, tq = 0.f;
        for (int c = 0; c < 16; ++c) { ts += rsum[tid][c]; tq += rsq[tid][c]; }
        float mean = ts / 512.0f;
        float var  = (tq - 512.0f * mean * mean) / 511.0f;
        smean[tid] = mean; sinv[tid] = rsqrtf(var + 1e-5f);
    }
    __syncthreads();
    float mean = smean[tk], inv = sinv[tk];
    int t = t0 + tk;
    _Float16* op = out + ((size_t)bidx * 16000 + t) * 512 + cg * 32;
#pragma unroll
    for (int ch = 0; ch < 4; ++ch) {
        v8h hv;
#pragma unroll
        for (int j = 0; j < 8; ++j) {
            int m = cg * 32 + ch * 8 + j;
            float v = (hs[m][tk] - mean) * inv * nw[m] + nb[m];
            hv[j] = (_Float16)gelu_exact(v);
        }
        *(v8h*)(op + ch * 8) = hv;
    }
}

// ---------------------------------------------------------------------------
// Conv layers 2..7 as WMMA GEMM. Activations token-major f16 [B][L][512].
// Weights pre-permuted f16: W[m][dk*512+ci]. K chunked by 128 through LDS.
// Fused channel-LN(ddof=1)+GELU. Grid: (ceil(Lout/16), B), block 256.
// ---------------------------------------------------------------------------
__global__ void conv_gemm_ln_gelu(const _Float16* __restrict__ in,
                                  const _Float16* __restrict__ w,
                                  const float* __restrict__ bias,
                                  const float* __restrict__ nw,
                                  const float* __restrict__ nb,
                                  _Float16* __restrict__ out,
                                  int k, int s, int Lin, int Lout) {
    const int Ktot = 512 * k;
    const int bidx = blockIdx.y;
    const int t0   = blockIdx.x * 16;
    const int tid  = threadIdx.x;
    const int lane = tid & 31;
    const int wv   = tid >> 5;
    const int hl   = lane >> 4;
    const int ln16 = lane & 15;

    __shared__ _Float16 xs[16][136];    // token-major K-panel, padded
    __shared__ float hs[512][16];
    __shared__ float rsum[16][16], rsq[16][16];
    __shared__ float smean[16], sinv[16];

    v8f acc[4] = {};
    const _Float16* inb = in + (size_t)bidx * Lin * 512;

    // loader mapping: thread -> 8 contiguous K of one token
    const int ltok = (tid * 8) >> 7;      // 0..15
    const int lkl  = (tid * 8) & 127;

    for (int kc = 0; kc < Ktot; kc += 128) {
        __syncthreads();
        {
            int dk = kc >> 9;             // uniform over the 128-chunk
            int ci0 = (kc & 511) + lkl;
            int t = t0 + ltok;
            int p = s * t - (k - 1) + dk;
            v8h val = {};
            if (t < Lout && p >= 0 && p < Lin)
                val = *(const v8h*)(inb + (size_t)p * 512 + ci0);
            *(v8h*)(&xs[ltok][lkl]) = val;
        }
        __syncthreads();
#pragma unroll
        for (int kk = 0; kk < 128; kk += 32) {
            int kb = kk + (hl ? 8 : 0);
            v16h bf = cmb(*(const v8h*)(&xs[ln16][kb]),
                          *(const v8h*)(&xs[ln16][kb + 16]));
            int o1 = hl ? 8 : 0, o2 = hl ? 24 : 16;
#pragma unroll
            for (int mt = 0; mt < 4; ++mt) {
                int m = (wv * 4 + mt) * 16 + ln16;
                const _Float16* wr = w + (size_t)m * Ktot + kc + kk;
                v16h af = cmb(*(const v8h*)(wr + o1), *(const v8h*)(wr + o2));
                acc[mt] = wmma_f16(af, bf, acc[mt]);
            }
        }
    }
    __syncthreads();
#pragma unroll
    for (int mt = 0; mt < 4; ++mt) {
        int mbase = (wv * 4 + mt) * 16 + (hl ? 8 : 0);
#pragma unroll
        for (int v = 0; v < 8; ++v) {
            int m = mbase + v;
            hs[m][ln16] = acc[mt][v] + bias[m];
        }
    }
    __syncthreads();
    const int tk = tid >> 4, cg = tid & 15;
    float ssum = 0.f, ssq = 0.f;
    for (int j = 0; j < 32; ++j) { float v = hs[cg * 32 + j][tk]; ssum += v; ssq += v * v; }
    rsum[tk][cg] = ssum; rsq[tk][cg] = ssq;
    __syncthreads();
    if (tid < 16) {
        float ts = 0.f, tq = 0.f;
        for (int c = 0; c < 16; ++c) { ts += rsum[tid][c]; tq += rsq[tid][c]; }
        float mean = ts / 512.0f;
        float var  = (tq - 512.0f * mean * mean) / 511.0f;
        smean[tid] = mean; sinv[tid] = rsqrtf(var + 1e-5f);
    }
    __syncthreads();
    float mean = smean[tk], inv = sinv[tk];
    int t = t0 + tk;
    if (t < Lout) {
        _Float16* op = out + ((size_t)bidx * Lout + t) * 512 + cg * 32;
#pragma unroll
        for (int ch = 0; ch < 4; ++ch) {
            v8h hv;
#pragma unroll
            for (int j = 0; j < 8; ++j) {
                int m = cg * 32 + ch * 8 + j;
                float v = (hs[m][tk] - mean) * inv * nw[m] + nb[m];
                hv[j] = (_Float16)gelu_exact(v);
            }
            *(v8h*)(op + ch * 8) = hv;
        }
    }
}

// ---------------------------------------------------------------------------
// Token GEMM: C[m,n] = sum_k A[m,k]*W[n,k] + bias[n]. A,W are f16 row-major.
// fuse: 0 none, 1 GELU, 2 +=res (f32). Output: outh (f16) if non-null, else
// outf (f32); omode 1 = upsample scatter into outf.
// Each wave -> two 16x16 C tiles along N. Grid: (M/16, N/256), block 256.
// ---------------------------------------------------------------------------
__global__ void gemm_wmma(const _Float16* __restrict__ A, const _Float16* __restrict__ W,
                          const float* __restrict__ bias, const float* __restrict__ res,
                          float* __restrict__ outf, _Float16* __restrict__ outh,
                          int M, int N, int K, int fuse, int omode, int L) {
    const int tid  = threadIdx.x;
    const int lane = tid & 31;
    const int wv   = tid >> 5;
    const int hl   = lane >> 4;
    const int ln16 = lane & 15;
    const int mt   = blockIdx.x;
    const int nt   = (blockIdx.y * 8 + wv) * 2;

    const int m = mt * 16 + ln16;
    const _Float16* ar  = A + (size_t)m * K;
    const _Float16* wr0 = W + ((size_t)(nt * 16 + ln16)) * K;
    const _Float16* wr1 = wr0 + (size_t)16 * K;

    v8f acc0 = {}, acc1 = {};
#pragma unroll 2
    for (int kk = 0; kk < K; kk += 32) {
        int ka  = kk + (hl ? 16 : 0);
        int kb2 = ka + 8;
        int kbB = kk + (hl ? 16 : 0);
        v16h af  = cmb(*(const v8h*)(ar + ka),   *(const v8h*)(ar + kb2));
        v16h bf0 = cmb(*(const v8h*)(wr0 + kbB), *(const v8h*)(wr0 + kbB + 8));
        v16h bf1 = cmb(*(const v8h*)(wr1 + kbB), *(const v8h*)(wr1 + kbB + 8));
        acc0 = wmma_f16(af, bf0, acc0);
        acc1 = wmma_f16(af, bf1, acc1);
    }
#pragma unroll
    for (int tix = 0; tix < 2; ++tix) {
        v8f acc = tix ? acc1 : acc0;
        int n = (nt + tix) * 16 + ln16;
        float bn = bias[n];
#pragma unroll
        for (int v = 0; v < 8; ++v) {
            int mm = mt * 16 + v + (hl ? 8 : 0);
            float val = acc[v] + bn;
            if (fuse == 1) val = gelu_exact(val);
            if (omode == 1) {
                int bb = mm / L, l = mm - bb * L;
                int dch = n >> 1, r = n & 1;
                outf[((size_t)(bb * (2 * L) + 2 * l + r)) * 768 + dch] = val;
            } else if (outh) {
                outh[(size_t)mm * N + n] = (_Float16)val;
            } else {
                size_t idx = (size_t)mm * N + n;
                if (fuse == 2) val += res[idx];
                outf[idx] = val;
            }
        }
    }
}

// ---------------------------------------------------------------------------
// Row LayerNorm over D=768; f32 in, f16 out. Grid: (M), block 256.
// ---------------------------------------------------------------------------
__global__ void layernorm_k(const float* __restrict__ x, const float* __restrict__ w,
                            const float* __restrict__ b, _Float16* __restrict__ y) {
    const int row = blockIdx.x;
    const int tid = threadIdx.x;
    const float* xr = x + (size_t)row * 768;
    float v0 = xr[tid], v1 = xr[tid + 256], v2 = xr[tid + 512];
    __shared__ float sb[256];
    sb[tid] = v0 + v1 + v2;
    __syncthreads();
    for (int o = 128; o > 0; o >>= 1) { if (tid < o) sb[tid] += sb[tid + o]; __syncthreads(); }
    float mean = sb[0] / 768.0f;
    __syncthreads();
    float d0 = v0 - mean, d1 = v1 - mean, d2 = v2 - mean;
    sb[tid] = d0 * d0 + d1 * d1 + d2 * d2;
    __syncthreads();
    for (int o = 128; o > 0; o >>= 1) { if (tid < o) sb[tid] += sb[tid + o]; __syncthreads(); }
    float inv = rsqrtf(sb[0] / 768.0f + 1e-5f);
    _Float16* yr = y + (size_t)row * 768;
    yr[tid]       = (_Float16)(d0 * inv * w[tid]       + b[tid]);
    yr[tid + 256] = (_Float16)(d1 * inv * w[tid + 256] + b[tid + 256]);
    yr[tid + 512] = (_Float16)(d2 * inv * w[tid + 512] + b[tid + 512]);
}

// ---------------------------------------------------------------------------
// Attention per (b,h): L=250 padded to 256, d=64. Q/K/V f16, out f16.
// Grid: (12, 8), block 256.
// ---------------------------------------------------------------------------
__global__ void attention_k(const _Float16* __restrict__ q, const _Float16* __restrict__ km,
                            const _Float16* __restrict__ vm, _Float16* __restrict__ o) {
    const int h    = blockIdx.x;
    const int bb   = blockIdx.y;
    const int tid  = threadIdx.x;
    const int lane = tid & 31;
    const int wv   = tid >> 5;
    const int hl   = lane >> 4;
    const int ln16 = lane & 15;
    const int L    = 250;
    const float slope = (h < 8) ? exp2f(-(float)(h + 1)) : exp2f(-(0.5f + (float)(h - 8)));
    __shared__ _Float16 ps[8][16][256];

    const _Float16* qh = q  + (size_t)bb * L * 768 + h * 64;
    const _Float16* kh = km + (size_t)bb * L * 768 + h * 64;
    const _Float16* vh = vm + (size_t)bb * L * 768 + h * 64;
    const v8f zero8 = {};
    const v8h zh = {};
    const v16h zf = cmb(zh, zh);

    for (int rt = wv; rt < 16; rt += 8) {
        const int m0 = rt * 16;
        v8f S[16];
#pragma unroll
        for (int i = 0; i < 16; ++i) S[i] = zero8;

        // ---- S = Q K^T ----
#pragma unroll
        for (int kk = 0; kk < 64; kk += 32) {
            int ka = kk + (hl ? 16 : 0), kb2 = ka + 8;
            int mq = m0 + ln16;
            const _Float16* qr = qh + (size_t)(mq < L ? mq : L - 1) * 768;
            v16h af = (mq < L) ? cmb(*(const v8h*)(qr + ka), *(const v8h*)(qr + kb2)) : zf;
            for (int nt2 = 0; nt2 < 16; ++nt2) {
                int nq = nt2 * 16 + ln16;
                const _Float16* kr = kh + (size_t)(nq < L ? nq : L - 1) * 768 + kk + (hl ? 16 : 0);
                v16h bf = (nq < L) ? cmb(*(const v8h*)kr, *(const v8h*)(kr + 8)) : zf;
                S[nt2] = wmma_f16(af, bf, S[nt2]);
            }
        }

        // ---- scale + ALiBi + causal + softmax; P -> LDS f16 ----
        const float scale = 0.125f;
#pragma unroll
        for (int v = 0; v < 8; ++v) {
            int mrow = m0 + v + (hl ? 8 : 0);
            float mx = -3.0e30f;
#pragma unroll
            for (int nt2 = 0; nt2 < 16; ++nt2) {
                int nq = nt2 * 16 + ln16;
                float sv = S[nt2][v] * scale + slope * (float)(nq - mrow);
                if (nq > mrow || nq >= L) sv = -1.0e30f;
                S[nt2][v] = sv;
                mx = fmaxf(mx, sv);
            }
            for (int off = 1; off < 16; off <<= 1) mx = fmaxf(mx, __shfl_xor(mx, off, 16));
            float sum = 0.f;
#pragma unroll
            for (int nt2 = 0; nt2 < 16; ++nt2) {
                float e = __expf(S[nt2][v] - mx);
                S[nt2][v] = e;
                sum += e;
            }
            for (int off = 1; off < 16; off <<= 1) sum += __shfl_xor(sum, off, 16);
            float inv = 1.0f / sum;
#pragma unroll
            for (int nt2 = 0; nt2 < 16; ++nt2)
                ps[wv][v + (hl ? 8 : 0)][nt2 * 16 + ln16] = (_Float16)(S[nt2][v] * inv);
        }

        // ---- O = P V ----
        v8f O[4];
#pragma unroll
        for (int i = 0; i < 4; ++i) O[i] = zero8;
#pragma unroll
        for (int kk = 0; kk < 256; kk += 32) {
            int ka = kk + (hl ? 16 : 0), kb2 = ka + 8;
            v16h af = cmb(*(const v8h*)(&ps[wv][ln16][ka]),
                          *(const v8h*)(&ps[wv][ln16][kb2]));
#pragma unroll
            for (int dt = 0; dt < 4; ++dt) {
                int dv = dt * 16 + ln16;
                int kbase = kk + (hl ? 16 : 0);
                v16h bf;
#pragma unroll
                for (int j = 0; j < 16; ++j) {
                    int kq = kbase + j;
                    bf[j] = (kq < L) ? vh[(size_t)kq * 768 + dv] : (_Float16)0.0f;
                }
                O[dt] = wmma_f16(af, bf, O[dt]);
            }
        }
#pragma unroll
        for (int dt = 0; dt < 4; ++dt) {
#pragma unroll
            for (int v = 0; v < 8; ++v) {
                int mrow = m0 + v + (hl ? 8 : 0);
                if (mrow < L) {
                    int dv = dt * 16 + ln16;
                    o[((size_t)bb * L + mrow) * 768 + h * 64 + dv] = (_Float16)O[dt][v];
                }
            }
        }
    }
}

// ---------------------------------------------------------------------------
// Host orchestration
// ---------------------------------------------------------------------------
extern "C" void kernel_launch(void* const* d_in, const int* in_sizes, int n_in,
                              void* d_out, int out_size, void* d_ws, size_t ws_size,
                              hipStream_t stream) {
    (void)in_sizes; (void)n_in; (void)out_size; (void)ws_size;
    const int L = 250, M = 8 * L;   // 2000 tokens

    char* base = (char*)d_ws;
    size_t off = 0;
    auto alloc = [&](size_t bytes) -> void* {
        void* p = base + off;
        off = (off + bytes + 255) & ~(size_t)255;
        return p;
    };
    _Float16* bufA = (_Float16*)alloc((size_t)8 * 512 * 16000 * 2);
    _Float16* bufB = (_Float16*)alloc((size_t)8 * 512 * 8000 * 2);
    float*    hid  = (float*)   alloc((size_t)M * 768 * 4);
    _Float16* t0b  = (_Float16*)alloc((size_t)M * 768 * 2);
    _Float16* qb   = (_Float16*)alloc((size_t)M * 768 * 2);
    _Float16* kb   = (_Float16*)alloc((size_t)M * 768 * 2);
    _Float16* vb   = (_Float16*)alloc((size_t)M * 768 * 2);
    _Float16* ab   = (_Float16*)alloc((size_t)M * 768 * 2);
    _Float16* m1b  = (_Float16*)alloc((size_t)M * 3072 * 2);

    auto cvt = [&](const void* src, _Float16* dst, int n) {
        cvt_f32_to_f16<<<(n + 2047) / 2048, 256, 0, stream>>>((const float*)src, dst, n);
    };

    // ---- weight conversion ----
    struct CC { int k, s, Lin, Lout; };
    const CC cc[6] = {{3,2,16000,8000},{3,2,8000,4000},{3,2,4000,2000},
                      {3,2,2000,1000},{2,2,1000,500},{2,2,500,250}};
    _Float16* convw16[6];
    for (int l = 0; l < 6; ++l) {
        int k = cc[l].k, n = 512 * 512 * k;
        convw16[l] = (_Float16*)alloc((size_t)n * 2);
        conv_w_perm_f16<<<(n + 255) / 256, 256, 0, stream>>>(
            (const float*)d_in[2 + 4 * (l + 1)], convw16[l], k);
    }
    _Float16* projW16 = (_Float16*)alloc((size_t)768 * 512 * 2);
    cvt(d_in[30], projW16, 768 * 512);
    _Float16* blkW16[4][6];
    for (int blk = 0; blk < 4; ++blk) {
        int p = 32 + 16 * blk;
        const int widx[6] = {p + 2, p + 4, p + 6, p + 8, p + 12, p + 14};
        const int wsz[6]  = {768 * 768, 768 * 768, 768 * 768, 768 * 768,
                             3072 * 768, 768 * 3072};
        for (int j = 0; j < 6; ++j) {
            blkW16[blk][j] = (_Float16*)alloc((size_t)wsz[j] * 2);
            cvt(d_in[widx[j]], blkW16[blk][j], wsz[j]);
        }
    }
    _Float16* upW16 = (_Float16*)alloc((size_t)1536 * 768 * 2);
    cvt(d_in[96], upW16, 1536 * 768);

    auto gemm = [&](const _Float16* A, const _Float16* W, const float* bias,
                    const float* res, float* outf, _Float16* outh,
                    int Mx, int Nx, int Kx, int fuse, int omode) {
        dim3 g(Mx / 16, Nx / 256);
        gemm_wmma<<<g, 256, 0, stream>>>(A, W, bias, res, outf, outh,
                                         Mx, Nx, Kx, fuse, omode, L);
    };

    // ---- frontend ----
    conv1_ln_gelu<<<dim3(1000, 8), 256, 0, stream>>>(
        (const float*)d_in[0], (const float*)d_in[2], (const float*)d_in[3],
        (const float*)d_in[4], (const float*)d_in[5], bufA);
    _Float16* src = bufA; _Float16* dst = bufB;
    for (int l = 0; l < 6; ++l) {
        int bidx = 2 + 4 * (l + 1);
        dim3 g((cc[l].Lout + 15) / 16, 8);
        conv_gemm_ln_gelu<<<g, 256, 0, stream>>>(
            src, convw16[l], (const float*)d_in[bidx + 1],
            (const float*)d_in[bidx + 2], (const float*)d_in[bidx + 3],
            dst, cc[l].k, cc[l].s, cc[l].Lin, cc[l].Lout);
        _Float16* tmp = src; src = dst; dst = tmp;
    }
    // feat = src: token-major f16 [2000, 512]

    // ---- projection ----
    gemm(src, projW16, (const float*)d_in[31], nullptr, hid, nullptr,
         M, 768, 512, 0, 0);

    // ---- transformer blocks ----
    for (int blk = 0; blk < 4; ++blk) {
        int p = 32 + 16 * blk;
        layernorm_k<<<M, 256, 0, stream>>>(hid, (const float*)d_in[p + 0],
                                           (const float*)d_in[p + 1], t0b);
        gemm(t0b, blkW16[blk][0], (const float*)d_in[p + 3], nullptr, nullptr, qb, M, 768, 768, 0, 0);
        gemm(t0b, blkW16[blk][1], (const float*)d_in[p + 5], nullptr, nullptr, kb, M, 768, 768, 0, 0);
        gemm(t0b, blkW16[blk][2], (const float*)d_in[p + 7], nullptr, nullptr, vb, M, 768, 768, 0, 0);
        attention_k<<<dim3(12, 8), 256, 0, stream>>>(qb, kb, vb, ab);
        gemm(ab, blkW16[blk][3], (const float*)d_in[p + 9], hid, hid, nullptr, M, 768, 768, 2, 0);
        layernorm_k<<<M, 256, 0, stream>>>(hid, (const float*)d_in[p + 10],
                                           (const float*)d_in[p + 11], t0b);
        gemm(t0b, blkW16[blk][4], (const float*)d_in[p + 13], nullptr, nullptr, m1b, M, 3072, 768, 1, 0);
        gemm(m1b, blkW16[blk][5], (const float*)d_in[p + 15], hid, hid, nullptr, M, 768, 3072, 2, 0);
    }

    // ---- upsample ----
    cvt_f32_to_f16<<<(M * 768 + 2047) / 2048, 256, 0, stream>>>(hid, t0b, M * 768);
    gemm(t0b, upW16, (const float*)d_in[97], nullptr, (float*)d_out, nullptr,
         M, 1536, 768, 0, 1);
}
